// SamplerModel_37967510896868
// MI455X (gfx1250) — matrix operand order repaired and checked by reading, not compiled, not run on hardware
//
#include <hip/hip_runtime.h>
#include <hip/hip_bf16.h>
#include <math.h>

// ---------------------------------------------------------------------------
// Problem constants (match reference): H=128, B=128, NA=32, N=4096, NH=4,
// DH=32, F=259 (padded to 288 = 9*32 for WMMA K-loop), QKV=384.
// ---------------------------------------------------------------------------
#define NATOMS 4096
#define HDIM   128
#define NHEADS 4
#define DHEAD  32
#define QKVDIM 384
#define FRAW   259
#define FPAD   288
#define NMOL   128

typedef __attribute__((ext_vector_type(16))) _Float16 v16h;
typedef __attribute__((ext_vector_type(8)))  _Float16 v8h;
typedef __attribute__((ext_vector_type(8)))  float    v8f;

// ---------------------------------------------------------------------------
// WMMA tile loaders (wave32, 16x16x32 f16 -> f32).
//
// A-matrix 16x32 f16 layout (ISA 7.12.2): lane<16 holds row=lane with
// K runs [0..8) and [16..24); lane>=16 holds row=lane-16 with K runs
// [8..16) and [24..32). Each run is 8 contiguous f16 = one 16B load.
//
// B-matrix 32x16 f16 layout: lane<16 holds column=lane, K runs [0..8) and
// [16..24); lane>=16 holds column=lane-16, K runs [8..16) and [24..32).
//
// C/D 16x16 f32 layout: VGPR r, lanes 0-15 -> M=r, lanes 16-31 -> M=8+r,
// N = lane&15.
// ---------------------------------------------------------------------------
__device__ __forceinline__ v16h load_a16(const _Float16* __restrict__ base,
                                         int ldk, int row0, int k0, int lane) {
  const int r    = lane & 15;
  const int koff = (lane >> 4) << 3;
  const _Float16* p = base + (size_t)(row0 + r) * ldk + k0 + koff;
  v8h lo = *reinterpret_cast<const v8h*>(p);
  v8h hi = *reinterpret_cast<const v8h*>(p + 16);
  v16h a;
#pragma unroll
  for (int i = 0; i < 8; ++i) { a[i] = lo[i]; a[i + 8] = hi[i]; }
  return a;
}

__device__ __forceinline__ v16h load_b16(const _Float16* __restrict__ base,
                                         int ldk, int n0, int k0, int lane) {
  const int c    = lane & 15;
  const int koff = (lane >> 4) << 3;
  const _Float16* p = base + (size_t)(n0 + c) * ldk + k0 + koff;
  v8h lo = *reinterpret_cast<const v8h*>(p);
  v8h hi = *reinterpret_cast<const v8h*>(p + 16);
  v16h b;
#pragma unroll
  for (int i = 0; i < 8; ++i) { b[i] = lo[i]; b[i + 8] = hi[i]; }
  return b;
}

// ---------------------------------------------------------------------------
// Generic WMMA GEMM: out[M,N] = A[M,K] (f16, row-major, lda) x Wp (f16 packed
// as [N, Kp] row-major) + bias. One wave per 16x16 tile. f32 and/or f16 out.
// ---------------------------------------------------------------------------
__global__ void k_gemm(const _Float16* __restrict__ A, int lda,
                       const _Float16* __restrict__ Wp, int Kp,
                       const float* __restrict__ bias,
                       float* __restrict__ outf, _Float16* __restrict__ outh,
                       int ldo, int ntilesN) {
  const int tile = blockIdx.x;
  const int m0   = (tile / ntilesN) * 16;
  const int n0   = (tile % ntilesN) * 16;
  const int lane = threadIdx.x;

  v8f acc = {};
  for (int k0 = 0; k0 < Kp; k0 += 32) {
    v16h a = load_a16(A, lda, m0, k0, lane);
    v16h b = load_b16(Wp, Kp, n0, k0, lane);
    acc = __builtin_amdgcn_wmma_f32_16x16x32_f16(false, a, false, b,
                                                 (short)0, acc, false, false);
  }

  const int n    = n0 + (lane & 15);
  const int mrow = m0 + ((lane >> 4) << 3);
  const float bv = bias ? bias[n] : 0.0f;
#pragma unroll
  for (int r = 0; r < 8; ++r) {
    float v = acc[r] + bv;
    if (outf) outf[(size_t)(mrow + r) * ldo + n] = v;
    if (outh) outh[(size_t)(mrow + r) * ldo + n] = (_Float16)v;
  }
}

// ---------------------------------------------------------------------------
// Flash attention: one wave per (head, 16-query tile). Online softmax over
// 4096 keys in 32-key chunks. Q/K read from qkv_h (f16, [4096, 384]:
// q | k | v sections), V read pre-transposed vT (f16, [128, 4096]).
// ---------------------------------------------------------------------------
__global__ void k_attn(const _Float16* __restrict__ qkv_h,
                       const _Float16* __restrict__ vT,
                       _Float16* __restrict__ o_h) {
  const int h    = blockIdx.y;
  const int m0   = blockIdx.x * 16;
  const int lane = threadIdx.x;
  const float scale = 0.17677669529663689f;  // 1/sqrt(32)

  __shared__ __align__(16) _Float16 S[16][32];

  // Q tile, A-layout (rows m0..m0+15, dh 0..31 of this head)
  v16h aq = load_a16(qkv_h, QKVDIM, m0, h * DHEAD, lane);

  v8f o0 = {}, o1 = {};
  float mstat[8], lstat[8];
#pragma unroll
  for (int r = 0; r < 8; ++r) { mstat[r] = -1e30f; lstat[r] = 0.0f; }

  for (int kb = 0; kb < NATOMS; kb += 32) {
    // K^T tiles: B[k=dh][n=key] = K[key][dh]
    v16h bk0 = load_b16(qkv_h + HDIM, QKVDIM, kb,      h * DHEAD, lane);
    v16h bk1 = load_b16(qkv_h + HDIM, QKVDIM, kb + 16, h * DHEAD, lane);
    v8f zero = {};
    v8f s0 = __builtin_amdgcn_wmma_f32_16x16x32_f16(false, aq, false, bk0,
                                                    (short)0, zero, false, false);
    v8f s1 = __builtin_amdgcn_wmma_f32_16x16x32_f16(false, aq, false, bk1,
                                                    (short)0, zero, false, false);

    // online softmax per query row (rows live in a 16-lane N-group)
#pragma unroll
    for (int r = 0; r < 8; ++r) {
      float a0 = s0[r] * scale, a1 = s1[r] * scale;
      float mx = fmaxf(a0, a1);
#pragma unroll
      for (int off = 1; off < 16; off <<= 1) mx = fmaxf(mx, __shfl_xor(mx, off, 16));
      float mnew = fmaxf(mstat[r], mx);
      float corr = __expf(mstat[r] - mnew);
      float p0 = __expf(a0 - mnew);
      float p1 = __expf(a1 - mnew);
      float rs = p0 + p1;
#pragma unroll
      for (int off = 1; off < 16; off <<= 1) rs += __shfl_xor(rs, off, 16);
      lstat[r] = lstat[r] * corr + rs;
      mstat[r] = mnew;
      o0[r] *= corr;
      o1[r] *= corr;
      // stash P (D-layout) into LDS for A-layout reload
      const int row = r + ((lane >> 4) << 3);
      const int col = lane & 15;
      S[row][col]      = (_Float16)p0;
      S[row][col + 16] = (_Float16)p1;
    }
    __syncthreads();

    v16h ap  = load_a16(&S[0][0], 32, 0, 0, lane);
    // V tiles: B[k=key][n=dh] = vT[(h*32+dh)*4096 + key]
    v16h bv0 = load_b16(vT, NATOMS, h * DHEAD,      kb, lane);
    v16h bv1 = load_b16(vT, NATOMS, h * DHEAD + 16, kb, lane);
    o0 = __builtin_amdgcn_wmma_f32_16x16x32_f16(false, ap, false, bv0,
                                                (short)0, o0, false, false);
    o1 = __builtin_amdgcn_wmma_f32_16x16x32_f16(false, ap, false, bv1,
                                                (short)0, o1, false, false);
    __syncthreads();

    if (kb + 32 < NATOMS) {
      // CDNA5 prefetch of the next K chunk (global_prefetch_b8)
      __builtin_prefetch(qkv_h + (size_t)(kb + 32) * QKVDIM + HDIM + h * DHEAD, 0, 0);
    }
  }

#pragma unroll
  for (int r = 0; r < 8; ++r) {
    const float inv = 1.0f / lstat[r];
    const int row = m0 + r + ((lane >> 4) << 3);
    const int col = lane & 15;
    o_h[(size_t)row * HDIM + h * DHEAD + col]      = (_Float16)(o0[r] * inv);
    o_h[(size_t)row * HDIM + h * DHEAD + col + 16] = (_Float16)(o1[r] * inv);
  }
}

// ---------------------------------------------------------------------------
// Weight pack: f32 [K,N] -> f16 [N,Kp] (transposed, zero-padded K)
// ---------------------------------------------------------------------------
__global__ void k_pack_w(const float* __restrict__ w, _Float16* __restrict__ wp,
                         int K, int N, int Kp) {
  int idx = blockIdx.x * blockDim.x + threadIdx.x;
  if (idx >= N * Kp) return;
  int n = idx / Kp, k = idx % Kp;
  wp[idx] = (k < K) ? (_Float16)w[(size_t)k * N + n] : (_Float16)0.0f;
}

// V repack: vT[c][atom] = qkv_h[atom][256 + c]
__global__ void k_pack_v(const _Float16* __restrict__ qkv_h, _Float16* __restrict__ vT) {
  int idx = blockIdx.x * blockDim.x + threadIdx.x;  // NATOMS*HDIM
  int atom = idx >> 7, c = idx & 127;
  vT[(size_t)c * NATOMS + atom] = qkv_h[(size_t)atom * QKVDIM + 2 * HDIM + c];
}

// ---------------------------------------------------------------------------
// Embedding gather + input LayerNorm -> x_f16 [4096,128]
// ---------------------------------------------------------------------------
__global__ void k_embed_ln(const int* __restrict__ elems, const float* __restrict__ emb,
                           const float* __restrict__ g, const float* __restrict__ b,
                           _Float16* __restrict__ xh) {
  const int i = blockIdx.x, t = threadIdx.x;
  __shared__ float red[128];
  float v = emb[(size_t)elems[i] * HDIM + t];
  red[t] = v; __syncthreads();
  for (int s = 64; s > 0; s >>= 1) { if (t < s) red[t] += red[t + s]; __syncthreads(); }
  float mean = red[0] * (1.0f / HDIM); __syncthreads();
  float d = v - mean;
  red[t] = d * d; __syncthreads();
  for (int s = 64; s > 0; s >>= 1) { if (t < s) red[t] += red[t + s]; __syncthreads(); }
  float var = red[0] * (1.0f / HDIM);
  xh[(size_t)i * HDIM + t] = (_Float16)(d * rsqrtf(var + 1e-5f) * g[t] + b[t]);
}

// ---------------------------------------------------------------------------
// Rowwise LayerNorm (+optional residual add, +optional SiLU), cols==blockDim
// ---------------------------------------------------------------------------
__global__ void k_row_ln(const float* __restrict__ x, const float* __restrict__ xadd,
                         const float* __restrict__ g, const float* __restrict__ b,
                         float* __restrict__ outf, _Float16* __restrict__ outh,
                         int do_silu) {
  const int row = blockIdx.x, t = threadIdx.x, C = blockDim.x;
  __shared__ float red[128];
  float v = x[(size_t)row * C + t];
  if (xadd) v += xadd[(size_t)row * C + t];
  red[t] = v; __syncthreads();
  for (int s = C >> 1; s > 0; s >>= 1) { if (t < s) red[t] += red[t + s]; __syncthreads(); }
  float mean = red[0] / C; __syncthreads();
  float d = v - mean;
  red[t] = d * d; __syncthreads();
  for (int s = C >> 1; s > 0; s >>= 1) { if (t < s) red[t] += red[t + s]; __syncthreads(); }
  float var = red[0] / C;
  float y = d * rsqrtf(var + 1e-5f) * g[t] + b[t];
  if (do_silu) y = y / (1.0f + __expf(-y));
  if (outf) outf[(size_t)row * C + t] = y;
  if (outh) outh[(size_t)row * C + t] = (_Float16)y;
}

// ---------------------------------------------------------------------------
// Cell encoder (rows 0..127 only; reference indexes cf[bidx], bidx<128):
// cf = silu(cell @ ce_w1 + b1) @ ce_w2 + b2
// ---------------------------------------------------------------------------
__global__ void k_cell(const float* __restrict__ cell, const float* __restrict__ w1,
                       const float* __restrict__ b1, const float* __restrict__ w2,
                       const float* __restrict__ b2, float* __restrict__ cf) {
  const int row = blockIdx.x, t = threadIdx.x;
  __shared__ float h1[128];
  float c0 = cell[row * 3 + 0], c1 = cell[row * 3 + 1], c2 = cell[row * 3 + 2];
  float a = c0 * w1[t] + c1 * w1[128 + t] + c2 * w1[256 + t] + b1[t];
  a = a / (1.0f + __expf(-a));
  h1[t] = a; __syncthreads();
  float acc = b2[t];
  for (int k = 0; k < 128; ++k) acc += h1[k] * w2[k * 128 + t];
  cf[(size_t)row * 128 + t] = acc;
}

// ---------------------------------------------------------------------------
// Concat [attn_out | cf[bidx] | coord] (259) + LayerNorm -> f16 [4096, 288]
// (padded with zeros for the WMMA K-loop). Reduction via ds_add_f32 atomics.
// ---------------------------------------------------------------------------
__global__ void k_concat_ln(const float* __restrict__ y, const float* __restrict__ cf,
                            const float* __restrict__ coord, const float* __restrict__ g,
                            const float* __restrict__ b, _Float16* __restrict__ comb) {
  const int i = blockIdx.x, t = threadIdx.x;  // blockDim = FPAD = 288
  __shared__ float ssum, ssq;
  if (t == 0) { ssum = 0.0f; ssq = 0.0f; }
  __syncthreads();
  const int bidx = i >> 5;  // NA==32 atoms per molecule (uniform, per setup)
  float v = 0.0f;
  const bool valid = (t < FRAW);
  if (t < 128)       v = y[(size_t)i * 128 + t];
  else if (t < 256)  v = cf[(size_t)bidx * 128 + (t - 128)];
  else if (t < 259)  v = coord[(size_t)i * 3 + (t - 256)];
  if (valid) { atomicAdd(&ssum, v); atomicAdd(&ssq, v * v); }
  __syncthreads();
  float mean = ssum * (1.0f / FRAW);
  float var  = ssq * (1.0f / FRAW) - mean * mean;
  float r = valid ? ((v - mean) * rsqrtf(var + 1e-5f) * g[t] + b[t]) : 0.0f;
  comb[(size_t)i * FPAD + t] = (_Float16)r;
}

// ---------------------------------------------------------------------------
// Final head: tanh(h3 @ op_w3 + b3)*0.01 + coord, per-molecule clamp
// ---------------------------------------------------------------------------
__global__ void k_final(const float* __restrict__ h3, const float* __restrict__ w3,
                        const float* __restrict__ b3, const float* __restrict__ coord,
                        const float* __restrict__ cell, float* __restrict__ out) {
  const int i = blockIdx.x, t = threadIdx.x;
  __shared__ float sh[128];
  sh[t] = h3[(size_t)i * 128 + t];
  __syncthreads();
  if (t < 3) {
    float acc = b3[t];
    for (int k = 0; k < 128; ++k) acc += sh[k] * w3[k * 3 + t];
    float off = tanhf(acc) * 0.01f;
    const int mb = i >> 5;
    const float* cr = cell + (size_t)(mb * 3 + t) * 3;
    float cmin = fminf(cr[0], fminf(cr[1], cr[2])) + 1e-6f;
    float cmax = fmaxf(cr[0], fmaxf(cr[1], cr[2])) - 1e-6f;
    float p = coord[(size_t)i * 3 + t] + off;
    out[(size_t)i * 3 + t] = fminf(fmaxf(p, cmin), cmax);
  }
}

// ---------------------------------------------------------------------------
// Launch
// ---------------------------------------------------------------------------
extern "C" void kernel_launch(void* const* d_in, const int* in_sizes, int n_in,
                              void* d_out, int out_size, void* d_ws, size_t ws_size,
                              hipStream_t stream) {
  (void)in_sizes; (void)n_in; (void)out_size; (void)ws_size;
  // inputs (setup_inputs order)
  const int*   elems      = (const int*)d_in[1];
  const float* cell       = (const float*)d_in[2];
  const float* coord      = (const float*)d_in[3];
  const float* emb        = (const float*)d_in[4];
  const float* ln_in_g    = (const float*)d_in[5];
  const float* ln_in_b    = (const float*)d_in[6];
  const float* attn_in_w  = (const float*)d_in[7];
  const float* attn_in_b  = (const float*)d_in[8];
  const float* attn_out_w = (const float*)d_in[9];
  const float* attn_out_b = (const float*)d_in[10];
  const float* ce_w1      = (const float*)d_in[11];
  const float* ce_b1      = (const float*)d_in[12];
  const float* ce_w2      = (const float*)d_in[13];
  const float* ce_b2      = (const float*)d_in[14];
  const float* ln_feat_g  = (const float*)d_in[15];
  const float* ln_feat_b  = (const float*)d_in[16];
  const float* op_w1      = (const float*)d_in[17];
  const float* op_b1      = (const float*)d_in[18];
  const float* op_ln1_g   = (const float*)d_in[19];
  const float* op_ln1_b   = (const float*)d_in[20];
  const float* res_w1     = (const float*)d_in[21];
  const float* res_b1     = (const float*)d_in[22];
  const float* res_ln_g   = (const float*)d_in[23];
  const float* res_ln_b   = (const float*)d_in[24];
  const float* res_w2     = (const float*)d_in[25];
  const float* res_b2     = (const float*)d_in[26];
  const float* op_ln2_g   = (const float*)d_in[27];
  const float* op_ln2_b   = (const float*)d_in[28];
  const float* op_w3      = (const float*)d_in[29];
  const float* op_b3      = (const float*)d_in[30];
  float* out = (float*)d_out;

  // workspace bump allocator (256B aligned)
  char* wsp = (char*)d_ws;
  size_t off = 0;
  auto alloc = [&](size_t bytes) -> void* {
    void* p = wsp + off;
    off = (off + bytes + 255) & ~(size_t)255;
    return p;
  };

  _Float16* x_h    = (_Float16*)alloc((size_t)NATOMS * HDIM * 2);
  _Float16* qkv_h  = (_Float16*)alloc((size_t)NATOMS * QKVDIM * 2);
  _Float16* vT     = (_Float16*)alloc((size_t)HDIM * NATOMS * 2);
  _Float16* o_h    = (_Float16*)alloc((size_t)NATOMS * HDIM * 2);
  float*    y_f    = (float*)alloc((size_t)NATOMS * HDIM * 4);
  float*    cf     = (float*)alloc((size_t)NMOL * HDIM * 4);
  _Float16* comb_h = (_Float16*)alloc((size_t)NATOMS * FPAD * 2);
  float*    g1     = (float*)alloc((size_t)NATOMS * HDIM * 4);
  float*    h1f    = (float*)alloc((size_t)NATOMS * HDIM * 4);
  _Float16* h1h    = (_Float16*)alloc((size_t)NATOMS * HDIM * 2);
  float*    rbf    = (float*)alloc((size_t)NATOMS * HDIM * 4);
  _Float16* r2h    = (_Float16*)alloc((size_t)NATOMS * HDIM * 2);
  float*    r3f    = (float*)alloc((size_t)NATOMS * HDIM * 4);
  float*    h3f    = (float*)alloc((size_t)NATOMS * HDIM * 4);
  _Float16* wqkv_p = (_Float16*)alloc((size_t)QKVDIM * HDIM * 2);
  _Float16* wout_p = (_Float16*)alloc((size_t)HDIM * HDIM * 2);
  _Float16* wop1_p = (_Float16*)alloc((size_t)HDIM * FPAD * 2);
  _Float16* wr1_p  = (_Float16*)alloc((size_t)HDIM * HDIM * 2);
  _Float16* wr2_p  = (_Float16*)alloc((size_t)HDIM * HDIM * 2);

  // weight packing (f32 [K,N] -> f16 [N,Kp])
  k_pack_w<<<(QKVDIM * HDIM + 255) / 256, 256, 0, stream>>>(attn_in_w, wqkv_p, HDIM, QKVDIM, HDIM);
  k_pack_w<<<(HDIM * HDIM + 255) / 256, 256, 0, stream>>>(attn_out_w, wout_p, HDIM, HDIM, HDIM);
  k_pack_w<<<(HDIM * FPAD + 255) / 256, 256, 0, stream>>>(op_w1, wop1_p, FRAW, HDIM, FPAD);
  k_pack_w<<<(HDIM * HDIM + 255) / 256, 256, 0, stream>>>(res_w1, wr1_p, HDIM, HDIM, HDIM);
  k_pack_w<<<(HDIM * HDIM + 255) / 256, 256, 0, stream>>>(res_w2, wr2_p, HDIM, HDIM, HDIM);

  // x = LN(emb[elems])
  k_embed_ln<<<NATOMS, HDIM, 0, stream>>>(elems, emb, ln_in_g, ln_in_b, x_h);

  // qkv = x @ attn_in_w + b   (f16 out only; q/k read in place, v repacked)
  k_gemm<<<(NATOMS / 16) * (QKVDIM / 16), 32, 0, stream>>>(
      x_h, HDIM, wqkv_p, HDIM, attn_in_b, nullptr, qkv_h, QKVDIM, QKVDIM / 16);
  k_pack_v<<<(NATOMS * HDIM + 255) / 256, 256, 0, stream>>>(qkv_h, vT);

  // flash attention
  k_attn<<<dim3(NATOMS / 16, NHEADS), 32, 0, stream>>>(qkv_h, vT, o_h);

  // attn out projection
  k_gemm<<<(NATOMS / 16) * (HDIM / 16), 32, 0, stream>>>(
      o_h, HDIM, wout_p, HDIM, attn_out_b, y_f, nullptr, HDIM, HDIM / 16);

  // cell encoder + concat + feature LN
  k_cell<<<NMOL, HDIM, 0, stream>>>(cell, ce_w1, ce_b1, ce_w2, ce_b2, cf);
  k_concat_ln<<<NATOMS, FPAD, 0, stream>>>(y_f, cf, coord, ln_feat_g, ln_feat_b, comb_h);

  // offset predictor MLP
  k_gemm<<<(NATOMS / 16) * (HDIM / 16), 32, 0, stream>>>(
      comb_h, FPAD, wop1_p, FPAD, op_b1, g1, nullptr, HDIM, HDIM / 16);
  k_row_ln<<<NATOMS, HDIM, 0, stream>>>(g1, nullptr, op_ln1_g, op_ln1_b, h1f, h1h, 1);

  k_gemm<<<(NATOMS / 16) * (HDIM / 16), 32, 0, stream>>>(
      h1h, HDIM, wr1_p, HDIM, res_b1, rbf, nullptr, HDIM, HDIM / 16);
  k_row_ln<<<NATOMS, HDIM, 0, stream>>>(rbf, nullptr, res_ln_g, res_ln_b, nullptr, r2h, 1);
  k_gemm<<<(NATOMS / 16) * (HDIM / 16), 32, 0, stream>>>(
      r2h, HDIM, wr2_p, HDIM, res_b2, r3f, nullptr, HDIM, HDIM / 16);

  // h = h1 + residual, then LN
  k_row_ln<<<NATOMS, HDIM, 0, stream>>>(r3f, h1f, op_ln2_g, op_ln2_b, h3f, nullptr, 0);

  // final head + clamp
  k_final<<<NATOMS, HDIM, 0, stream>>>(h3f, op_w3, op_b3, coord, cell, out);
}
